// AlignCell_64553358458927
// MI455X (gfx1250) — compile-verified, matrix operand order checked by reading, not run on hardware
//
#include <hip/hip_runtime.h>
#include <hip/hip_bf16.h>

// ---------------------------------------------------------------------------
// AlignCell (attention-GRU) for MI455X / gfx1250 (wave32, WMMA, TDM).
//
// Dominant cost: context(64x2048x512) @ Wa(512x512) = 68.7 GFLOP, fused with
// tanh/Va reduction so the 268MB context streams from HBM exactly once for
// the score pass. All GEMMs use v_wmma_f32_16x16x32_bf16 (bf16 in, f32 acc).
// The hot kernel stages packed Wa fragments into LDS with the Tensor Data
// Mover (double-buffered, s_wait_tensorcnt + workgroup barriers) so the 8
// waves of a workgroup share one L2 fetch instead of eight.
// ---------------------------------------------------------------------------

typedef __bf16  bf16;
typedef __attribute__((ext_vector_type(16))) __bf16 v16bf;
typedef __attribute__((ext_vector_type(8)))  float  v8f;
typedef __attribute__((ext_vector_type(4)))  float  v4f;
typedef __attribute__((ext_vector_type(4)))  unsigned int u32x4;
typedef __attribute__((ext_vector_type(8)))  int    i32x8;
typedef __attribute__((ext_vector_type(4)))  int    i32x4;

#define UNITS 512
#define EMB   256
#define B     64
#define T     2048

#if __has_builtin(__builtin_amdgcn_tensor_load_to_lds)
#define HAVE_TDM 1
#else
#define HAVE_TDM 0
#endif

// ---- WMMA fragment conventions (wave32, 16x16x32 bf16) --------------------
// A (16xK tile, row-major source): lane L holds row m = L&15; khalf = (L>>4)*8;
//   vector elems 0..7  -> K = k0 + khalf + i
//   vector elems 8..15 -> K = k0 + 16 + khalf + (i-8)
// B (Kx16 tile): lane L holds col n = L&15, same K mapping as A.
// C/D (16x16 f32, 8 VGPRs): elem r -> row (L>>4)*8 + r, col L&15.

__device__ inline v8f wmma_bf16(v16bf a, v16bf b, v8f c) {
  return __builtin_amdgcn_wmma_f32_16x16x32_bf16(false, a, false, b,
                                                 (short)0, c, false, false);
}

#if HAVE_TDM
// TDM: DMA one contiguous 16KB chunk (4096 DWORDs) from global into LDS.
// D# per CDNA5 ISA: 1-D tile, data_size=4B (code 2),
// tensor_dim0 = tile_dim0 = 4096, tile_dim1 = 1, type=2 ("image").
__device__ inline void tdm_load_16k(const void* gsrc, unsigned lds_off) {
  unsigned long long ga = (unsigned long long)(uintptr_t)gsrc;
  u32x4 g0;
  g0[0] = 1u;                                           // count=1, user D#
  g0[1] = lds_off;                                      // lds_addr (bytes)
  g0[2] = (unsigned)(ga & 0xffffffffu);                 // global_addr[31:0]
  g0[3] = (unsigned)((ga >> 32) & 0x01ffffffu)          // global_addr[56:32]
          | (2u << 30);                                 // type = 2
  i32x8 g1 = {0, 0, 0, 0, 0, 0, 0, 0};
  g1[0] = (2 << 16);                 // workgroup_mask=0, data_size=4B
  g1[1] = (int)(4096u << 16);        // tensor_dim0[15:0] = 4096
  g1[2] = (int)(1u << 16);           // tensor_dim0 hi=0, tensor_dim1 = 1
  g1[3] = (int)(4096u << 16);        // tensor_dim1 hi=0, tile_dim0 = 4096
  g1[4] = 1;                         // tile_dim1 = 1, tile_dim2 = 0
  g1[5] = 4096;                      // tensor_dim0_stride = 4096
  i32x4 z4 = {0, 0, 0, 0};
#if defined(__clang_major__) && (__clang_major__ >= 23)
  i32x8 z8 = {0, 0, 0, 0, 0, 0, 0, 0};
  __builtin_amdgcn_tensor_load_to_lds(g0, g1, z4, z4, z8, 0);
#else
  __builtin_amdgcn_tensor_load_to_lds(g0, g1, z4, z4, 0);
#endif
}
#endif // HAVE_TDM

// ---------------------------------------------------------------------------
// Pack a (K x N) f32 matrix into bf16 WMMA-B fragments.
// frag id = (n0/16)*(K/32) + (k0/32); each frag = 512 bf16 (lane*16 layout)
// so the hot loop loads one contiguous 32B v16bf per lane per K-step.
// For Wa (K=512): fragments for one 16-col block (nb) are 16 consecutive
// frags = 16KB contiguous -> exactly one TDM chunk.
// ---------------------------------------------------------------------------
__global__ void pack_b_kernel(const float* __restrict__ Bm, int ldb,
                              bf16* __restrict__ out, int K) {
  int lane = threadIdx.x & 31;
  int frag = blockIdx.x * (blockDim.x >> 5) + (threadIdx.x >> 5);
  int kfrags = K >> 5;
  int nb = frag / kfrags, kb = frag % kfrags;
  int col = (nb << 4) + (lane & 15);
  int k0  = (kb << 5) + ((lane >> 4) << 3);
  v16bf v;
#pragma unroll
  for (int i = 0; i < 8; ++i) v[i]     = (bf16)Bm[(size_t)(k0 + i) * ldb + col];
#pragma unroll
  for (int i = 0; i < 8; ++i) v[8 + i] = (bf16)Bm[(size_t)(k0 + 16 + i) * ldb + col];
  *(v16bf*)(out + ((size_t)frag << 9) + (lane << 4)) = v;
}

// ---------------------------------------------------------------------------
// Generic small GEMM: C[M,N] = A[M,K] @ B[K,N] (+ bias). One 16x16 tile per
// wave, K stepped by 32 with bf16 WMMA / f32 accumulation. A row-major f32,
// B row-major f32 (strided gather; these GEMMs are tiny, <0.4 GFLOP total).
// ---------------------------------------------------------------------------
__global__ void gemm_bf16_wmma(const float* __restrict__ A, int lda,
                               const float* __restrict__ Bm, int ldb,
                               const float* __restrict__ bias,
                               float* __restrict__ C, int ldc,
                               int M, int N, int K) {
  int lane = threadIdx.x & 31;
  int tile = blockIdx.x * (blockDim.x >> 5) + (threadIdx.x >> 5);
  int ntn  = N >> 4;
  if (tile >= (M >> 4) * ntn) return;          // whole-wave exit only
  int m0 = (tile / ntn) << 4;
  int n0 = (tile % ntn) << 4;
  int row   = m0 + (lane & 15);
  int col   = n0 + (lane & 15);
  int khalf = (lane >> 4) << 3;                // 0 or 8

  v8f c = {};
  for (int k0 = 0; k0 < K; k0 += 32) {
    v16bf a, b;
    const float* pa = A + (size_t)row * lda + k0 + khalf;
#pragma unroll
    for (int i = 0; i < 8; ++i) a[i]     = (bf16)pa[i];
#pragma unroll
    for (int i = 0; i < 8; ++i) a[8 + i] = (bf16)pa[16 + i];
    const float* pb = Bm + (size_t)(k0 + khalf) * ldb + col;
#pragma unroll
    for (int i = 0; i < 8; ++i) b[i]     = (bf16)pb[(size_t)i * ldb];
#pragma unroll
    for (int i = 0; i < 8; ++i) b[8 + i] = (bf16)pb[(size_t)(16 + i) * ldb];
    c = wmma_bf16(a, b, c);
  }
  float bv = bias ? bias[col] : 0.0f;
  int rbase = m0 + ((lane >> 4) << 3);
#pragma unroll
  for (int r = 0; r < 8; ++r)
    C[(size_t)(rbase + r) * ldc + col] = c[r] + bv;
}

// ---------------------------------------------------------------------------
// Fused attention-score kernel (the 68.7 GFLOP hot loop).
// Each wave: one 16-row tile of context (16 x 512). All 16 K-fragments of A
// live in VGPRs; loop 32 N-blocks x 16 WMMA each. Packed Wa fragments are
// TDM-staged into a 2x16KB LDS double buffer by wave 0 (one L2 fetch per
// workgroup instead of eight), pipeline: wait tensorcnt -> barrier ->
// issue next TDM -> compute -> barrier. Fuses +ba_w +hUa, tanh, *Va and the
// per-row column reduction -> score[b,t]. Context is read exactly once.
// ---------------------------------------------------------------------------
__global__ void __launch_bounds__(256)
attn_score_wmma(const float* __restrict__ ctx,     // (B, T, UNITS)
                const bf16*  __restrict__ WaP,     // packed Wa fragments
                const float* __restrict__ hUa,     // (B, UNITS) incl. ba_u
                const float* __restrict__ ba_w,    // (UNITS)
                const float* __restrict__ Va,      // (UNITS)
                const float* __restrict__ ba_v,    // (1)
                float*       __restrict__ score) { // (B, T)
#if HAVE_TDM
  __shared__ __align__(32) bf16 smem[2 * 8192];    // 2 x 16KB Wa chunks
#endif
  int lane = threadIdx.x & 31;
  int tile = blockIdx.x * (blockDim.x >> 5) + (threadIdx.x >> 5); // 0..8191
  int m0   = tile << 4;              // global row in (B*T)
  int bidx = m0 >> 11;               // batch = m0 / T
  int half = lane >> 4;

#if HAVE_TDM
  bool leader = (threadIdx.x < 32);  // wave 0 drives the TDM pipeline
  unsigned lds_base = (unsigned)(uintptr_t)(&smem[0]);
  if (leader) tdm_load_16k(WaP, lds_base);         // preload nb = 0
#endif

  // Load all 16 A fragments (K = 512) once; 32B-aligned v4f loads.
  const float* pa = ctx + (size_t)(m0 + (lane & 15)) * UNITS;
  v16bf afrag[16];
#pragma unroll
  for (int kk = 0; kk < 16; ++kk) {
    const v4f* p = (const v4f*)(pa + kk * 32 + half * 8);
    v4f f0 = p[0], f1 = p[1], f2 = p[4], f3 = p[5];
#pragma unroll
    for (int i = 0; i < 4; ++i) {
      afrag[kk][i]      = (bf16)f0[i];
      afrag[kk][4 + i]  = (bf16)f1[i];
      afrag[kk][8 + i]  = (bf16)f2[i];
      afrag[kk][12 + i] = (bf16)f3[i];
    }
  }

  float acc[8] = {0.f, 0.f, 0.f, 0.f, 0.f, 0.f, 0.f, 0.f};
  for (int nb = 0; nb < 32; ++nb) {
    int col   = (nb << 4) + (lane & 15);
    float add = hUa[(bidx << 9) + col] + ba_w[col];
    float va  = Va[col];
    v8f c = {};
#if HAVE_TDM
    if (leader) __builtin_amdgcn_s_wait_tensorcnt((short)0); // chunk nb landed
    __syncthreads();                                         // publish buffer
    if (leader && nb < 31)                                   // prefetch nb+1
      tdm_load_16k(WaP + ((size_t)(nb + 1) << 13),
                   lds_base + (unsigned)(((nb + 1) & 1) << 14));
    const bf16* bp = smem + ((nb & 1) << 13) + (lane << 4);
#pragma unroll
    for (int kk = 0; kk < 16; ++kk)
      c = wmma_bf16(afrag[kk], *(const v16bf*)(bp + (kk << 9)), c);
#else
    const bf16* bp = WaP + ((size_t)(nb << 4) << 9) + (lane << 4);
#pragma unroll
    for (int kk = 0; kk < 16; ++kk)
      c = wmma_bf16(afrag[kk], *(const v16bf*)(bp + ((size_t)kk << 9)), c);
#endif
#pragma unroll
    for (int r = 0; r < 8; ++r)
      acc[r] += tanhf(c[r] + add) * va;     // fold score @ Va on the fly
#if HAVE_TDM
    __syncthreads();                        // all waves done with buffer nb&1
#endif
  }

  // Butterfly-reduce each row's partial over the 16 column lanes.
#pragma unroll
  for (int r = 0; r < 8; ++r) {
    float v = acc[r];
    v += __shfl_xor(v, 8, 16);
    v += __shfl_xor(v, 4, 16);
    v += __shfl_xor(v, 2, 16);
    v += __shfl_xor(v, 1, 16);
    acc[r] = v;
  }
  if ((lane & 15) == 0) {
    float bav = ba_v[0];
    int t0 = (m0 & (T - 1)) + (half << 3);
#pragma unroll
    for (int r = 0; r < 8; ++r)
      score[((size_t)bidx << 11) + t0 + r] = acc[r] + bav;
  }
}

// ---------------------------------------------------------------------------
// Softmax over t (2048) per batch row; LDS tree reductions.
// ---------------------------------------------------------------------------
__global__ void softmax_kernel(const float* __restrict__ score,
                               float* __restrict__ attn) {
  __shared__ float red[256];
  int b = blockIdx.x, tid = threadIdx.x;
  const float* s = score + ((size_t)b << 11);
  float m = -3.4e38f;
  for (int t = tid; t < T; t += 256) m = fmaxf(m, s[t]);
  red[tid] = m; __syncthreads();
  for (int off = 128; off > 0; off >>= 1) {
    if (tid < off) red[tid] = fmaxf(red[tid], red[tid + off]);
    __syncthreads();
  }
  float bmax = red[0]; __syncthreads();
  float sum = 0.f;
  for (int t = tid; t < T; t += 256) sum += __expf(s[t] - bmax);
  red[tid] = sum; __syncthreads();
  for (int off = 128; off > 0; off >>= 1) {
    if (tid < off) red[tid] += red[tid + off];
    __syncthreads();
  }
  float inv = 1.0f / red[0];
  float* a = attn + ((size_t)b << 11);
  for (int t = tid; t < T; t += 256) a[t] = __expf(s[t] - bmax) * inv;
}

// ---------------------------------------------------------------------------
// ctx_vec[b,n] = sum_t attn[b,t] * context[b,t,n]; coalesced in n,
// second (and last) streaming pass over context; prefetch ahead.
// ---------------------------------------------------------------------------
__global__ void ctx_vec_kernel(const float* __restrict__ attn,
                               const float* __restrict__ ctx,
                               float* __restrict__ out) {
  int b = blockIdx.x >> 1;
  int n = ((blockIdx.x & 1) << 8) + threadIdx.x;
  const float* cb = ctx + ((size_t)b << 11) * UNITS + n;
  const float* ab = attn + ((size_t)b << 11);
  float acc = 0.f;
  for (int t = 0; t < T; t += 8) {
    __builtin_prefetch(cb + (size_t)(t + 32) * UNITS, 0, 1);
#pragma unroll
    for (int u = 0; u < 8; ++u)
      acc += ab[t + u] * cb[(size_t)(t + u) * UNITS];
  }
  out[((size_t)b << 9) + n] = acc;
}

// ---------------------------------------------------------------------------
// Gate kernels (elementwise).
// ---------------------------------------------------------------------------
__global__ void gates_kernel(const float* __restrict__ xg,
                             const float* __restrict__ rec,
                             const float* __restrict__ cg,
                             const float* __restrict__ h_tm1,
                             float* __restrict__ z_out,
                             float* __restrict__ rh_out) {
  int i = blockIdx.x * 256 + threadIdx.x;        // 0 .. B*UNITS-1
  int b = i >> 9, n = i & (UNITS - 1);
  const float* xr = xg + (size_t)b * 3 * UNITS;
  const float* cr = cg + (size_t)b * 3 * UNITS;
  float zs = xr[n]         + rec[(size_t)b * 2 * UNITS + n]         + cr[n];
  float rs = xr[UNITS + n] + rec[(size_t)b * 2 * UNITS + UNITS + n] + cr[UNITS + n];
  float z = 1.f / (1.f + __expf(-zs));
  float r = 1.f / (1.f + __expf(-rs));
  z_out[i]  = z;
  rh_out[i] = r * h_tm1[i];
}

__global__ void newh_kernel(const float* __restrict__ xg,
                            const float* __restrict__ cg,
                            const float* __restrict__ rec_h,
                            const float* __restrict__ z,
                            const float* __restrict__ h_tm1,
                            float* __restrict__ h_ws,
                            float* __restrict__ h_out) {
  int i = blockIdx.x * 256 + threadIdx.x;
  int b = i >> 9, n = i & (UNITS - 1);
  float xh = xg[(size_t)b * 3 * UNITS + 2 * UNITS + n];
  float ch = cg[(size_t)b * 3 * UNITS + 2 * UNITS + n];
  float hb = tanhf(xh + rec_h[i] + ch);
  float zz = z[i];
  float h  = zz * h_tm1[i] + (1.f - zz) * hb;
  h_ws[i]  = h;
  h_out[i] = h;
}

// ---------------------------------------------------------------------------
extern "C" void kernel_launch(void* const* d_in, const int* in_sizes, int n_in,
                              void* d_out, int out_size, void* d_ws, size_t ws_size,
                              hipStream_t stream) {
  const float* inputs   = (const float*)d_in[0];   // (64,256)
  const float* h_tm1    = (const float*)d_in[1];   // (64,512)
  const float* context  = (const float*)d_in[2];   // (64,2048,512)
  const float* Wi       = (const float*)d_in[3];   // (256,512)
  const float* bi       = (const float*)d_in[4];
  const float* kern     = (const float*)d_in[5];   // (512,1536)
  const float* rec_k    = (const float*)d_in[6];   // (512,1536)
  const float* attn_k   = (const float*)d_in[7];   // (512,1536)
  const float* bias     = (const float*)d_in[8];
  const float* Wa       = (const float*)d_in[9];   // (512,512)
  const float* ba_w     = (const float*)d_in[10];
  const float* Ua       = (const float*)d_in[11];  // (512,512)
  const float* ba_u     = (const float*)d_in[12];
  const float* Va       = (const float*)d_in[13];  // (512,1)
  const float* ba_v     = (const float*)d_in[14];  // (1)
  const float* Wo       = (const float*)d_in[15];  // (512,512)
  const float* bo       = (const float*)d_in[16];

  float* out_p = (float*)d_out;                    // out (64,512) then h (64,512)
  float* h_out = out_p + B * UNITS;

  // Workspace carve-up (floats), all offsets 32B-aligned.
  float* W     = (float*)d_ws;
  float* x     = W;                 // 64*512
  float* xg    = x    + 32768;      // 64*1536
  float* hUa   = xg   + 98304;      // 64*512
  float* rec   = hUa  + 32768;      // 64*1024
  float* score = rec  + 65536;      // 64*2048
  float* attn  = score+ 131072;     // 64*2048
  float* cvec  = attn + 131072;     // 64*512
  float* cg    = cvec + 32768;      // 64*1536
  float* zbuf  = cg   + 98304;      // 64*512
  float* rh    = zbuf + 32768;      // 64*512
  float* rech  = rh   + 32768;      // 64*512
  float* hws   = rech + 32768;      // 64*512
  bf16*  WaP   = (bf16*)(hws + 32768); // 512*512 bf16 fragments (512 KB)

  // 0) pack Wa into WMMA-B fragment order (512 frags, 8 waves/block)
  pack_b_kernel<<<64, 256, 0, stream>>>(Wa, UNITS, WaP, UNITS);

  // 1) x = inputs @ Wi + bi            (64x512, K=256) -> 128 tiles
  gemm_bf16_wmma<<<16, 256, 0, stream>>>(inputs, EMB, Wi, UNITS, bi,
                                         x, UNITS, B, UNITS, EMB);
  // 2) xg = x @ kernel + bias          (64x1536, K=512) -> 384 tiles
  gemm_bf16_wmma<<<48, 256, 0, stream>>>(x, UNITS, kern, 3 * UNITS, bias,
                                         xg, 3 * UNITS, B, 3 * UNITS, UNITS);
  // 3) hUa = h_tm1 @ Ua + ba_u         (64x512) -> 128 tiles
  gemm_bf16_wmma<<<16, 256, 0, stream>>>(h_tm1, UNITS, Ua, UNITS, ba_u,
                                         hUa, UNITS, B, UNITS, UNITS);
  // 4) rec[z|r] = h_tm1 @ rec_k[:, :1024]  (64x1024) -> 256 tiles
  gemm_bf16_wmma<<<32, 256, 0, stream>>>(h_tm1, UNITS, rec_k, 3 * UNITS, nullptr,
                                         rec, 2 * UNITS, B, 2 * UNITS, UNITS);
  // 5) fused score: tanh(ctx@Wa + ba_w + hUa) @ Va + ba_v  (8192 tiles)
  attn_score_wmma<<<1024, 256, 0, stream>>>(context, WaP, hUa, ba_w, Va, ba_v,
                                            score);
  // 6) softmax over t
  softmax_kernel<<<B, 256, 0, stream>>>(score, attn);
  // 7) ctx_vec = sum_t attn * context
  ctx_vec_kernel<<<2 * B, 256, 0, stream>>>(attn, context, cvec);
  // 8) cg = ctx_vec @ attention_kernel (no bias) -> 384 tiles
  gemm_bf16_wmma<<<48, 256, 0, stream>>>(cvec, UNITS, attn_k, 3 * UNITS, nullptr,
                                         cg, 3 * UNITS, B, 3 * UNITS, UNITS);
  // 9) z, r*h_tm1
  gates_kernel<<<128, 256, 0, stream>>>(xg, rec, cg, h_tm1, zbuf, rh);
  // 10) rec_h = (r*h) @ rec_k[:, 1024:1536] -> 128 tiles
  gemm_bf16_wmma<<<16, 256, 0, stream>>>(rh, UNITS, rec_k + 2 * UNITS, 3 * UNITS,
                                         nullptr, rech, UNITS, B, UNITS, UNITS);
  // 11) h = z*h_tm1 + (1-z)*tanh(x_h + rec_h + c_h); also second output
  newh_kernel<<<128, 256, 0, stream>>>(xg, cg, rech, zbuf, h_tm1, hws, h_out);
  // 12) out = h @ Wo + bo -> first output
  gemm_bf16_wmma<<<16, 256, 0, stream>>>(hws, UNITS, Wo, UNITS, bo,
                                         out_p, UNITS, B, UNITS, UNITS);
}
